// GeneMamba2_31722628448801
// MI455X (gfx1250) — compile-verified, matrix operand order checked by reading, not run on hardware
//
#include <hip/hip_runtime.h>
#include <hip/hip_bf16.h>
#include <math.h>

// ---------------- model dims ----------------
#define D_MODEL   512
#define D_STATE   64
#define D_CONV    4
#define D_INNER   1024
#define NHEADS    16
#define HEADDIM   64
#define D_IN_PROJ 2304      // 2*D_INNER + 2*D_STATE + NHEADS = 2192 cols read, padded to 9*256
#define CONV_DIM  1152      // D_INNER + 2*D_STATE
#define N_LAYERS  2
#define VOCABSZ   32768
#define BATCH     2
#define SEQLEN    512
#define BL        (BATCH * SEQLEN)   // 1024 rows
#define EPSF      1e-5f

// ---------------- bf16 / wmma types ----------------
typedef __attribute__((ext_vector_type(16))) __bf16 v16bf;
typedef __attribute__((ext_vector_type(8)))  __bf16 v8bf;
typedef __attribute__((ext_vector_type(8)))  float  v8f;

union BfFrag { v16bf v; v8bf h[2]; };

__device__ __forceinline__ __bf16 f32_to_bf16(float f) {
  unsigned u = __float_as_uint(f);
  unsigned r = (u + 0x7FFFu + ((u >> 16) & 1u)) >> 16;
  unsigned short s = (unsigned short)r;
  return __builtin_bit_cast(__bf16, s);
}

__device__ __forceinline__ float siluf(float x) { return x / (1.0f + __expf(-x)); }
__device__ __forceinline__ float softplusf(float x) {
  return (x > 20.0f) ? x : log1pf(__expf(x));
}
__device__ __forceinline__ float sigmf(float x) { return 1.0f / (1.0f + __expf(-x)); }

// ============================================================
// bf16 WMMA GEMM:  C[M,N] = A[M,K] * W[N,K]^T (+bias[n]) (+addend[m,n])
// A, W row-major bf16.  K is a compile-time constant; the K loop is fully
// unrolled with an explicit 2-deep software pipeline: the 10 b128 loads for
// K-step s+1 are issued before the 4 WMMAs of step s, with independent
// ping-pong register sets so the scheduler can use partial s_wait_loadcnt.
// __launch_bounds__(256, 1) lifts the 64-VGPR occupancy cap that otherwise
// forces single-register-set serialization.
// Block = 256 threads = 8 waves arranged 2(M) x 4(N); each wave computes a
// 16x64 C strip (4 accumulators), reusing its A fragment 4x.
// Requires M%32==0, N%256==0, K%32==0.
// ============================================================
#define GEMM_BM 32
#define GEMM_BN 256

struct Frags { BfFrag a, b0, b1, b2, b3; };

__device__ __forceinline__ void load_step(Frags& f,
                                          const __bf16* __restrict__ a_row,
                                          const __bf16* __restrict__ w0,
                                          const __bf16* __restrict__ w1,
                                          const __bf16* __restrict__ w2,
                                          const __bf16* __restrict__ w3,
                                          int k0) {
  f.a.h[0]  = *reinterpret_cast<const v8bf*>(a_row + k0);
  f.a.h[1]  = *reinterpret_cast<const v8bf*>(a_row + k0 + 16);
  f.b0.h[0] = *reinterpret_cast<const v8bf*>(w0 + k0);
  f.b0.h[1] = *reinterpret_cast<const v8bf*>(w0 + k0 + 16);
  f.b1.h[0] = *reinterpret_cast<const v8bf*>(w1 + k0);
  f.b1.h[1] = *reinterpret_cast<const v8bf*>(w1 + k0 + 16);
  f.b2.h[0] = *reinterpret_cast<const v8bf*>(w2 + k0);
  f.b2.h[1] = *reinterpret_cast<const v8bf*>(w2 + k0 + 16);
  f.b3.h[0] = *reinterpret_cast<const v8bf*>(w3 + k0);
  f.b3.h[1] = *reinterpret_cast<const v8bf*>(w3 + k0 + 16);
}

template <int K>
__global__ __launch_bounds__(256, 1)
void gemm_bf16_wmma(const __bf16* __restrict__ A,
                    const __bf16* __restrict__ W,
                    float* __restrict__ C,
                    const float* __restrict__ bias,    // len N or null
                    const float* __restrict__ addend,  // M*N or null
                    int M, int N)
{
  const int wave = threadIdx.x >> 5;
  const int lane = threadIdx.x & 31;
  const int wm = wave >> 2;          // 0..1
  const int wn = wave & 3;           // 0..3
  const int m0 = blockIdx.x * GEMM_BM + wm * 16;
  const int n0 = blockIdx.y * GEMM_BN + wn * 64;
  if (m0 >= M || n0 >= N) return;    // wave-uniform -> EXEC all-ones at every WMMA

  const int rl = lane & 15;          // tile row (A) / tile col-row (W)
  const int hi = (lane >> 4) & 1;    // K sub-pattern group

  // per-lane base pointers; all K-step addressing below is immediate offsets
  const __bf16* a_row = A + (size_t)(m0 + rl) * K + hi * 8;
  const __bf16* w0 = W + (size_t)(n0 + rl) * K + hi * 8;   // N-tile 0
  const __bf16* w1 = w0 + (size_t)16 * K;                  // N-tile 1
  const __bf16* w2 = w0 + (size_t)32 * K;                  // N-tile 2
  const __bf16* w3 = w0 + (size_t)48 * K;                  // N-tile 3

  v8f acc0 = {}, acc1 = {}, acc2 = {}, acc3 = {};

  Frags f[2];
  load_step(f[0], a_row, w0, w1, w2, w3, 0);
#pragma unroll
  for (int k0 = 0; k0 < K; k0 += 32) {
    const int s = (k0 >> 5) & 1;     // constant after full unroll
    if (k0 + 32 < K) {
      // issue next step's loads into the other register set before this
      // step's WMMAs so waits can be partial
      load_step(f[s ^ 1], a_row, w0, w1, w2, w3, k0 + 32);
      // prefetch streaming weight rows 2 K-steps ahead (global_prefetch_b8)
      if (k0 + 64 < K) {
        __builtin_prefetch(w0 + k0 + 64, 0, 1);
        __builtin_prefetch(w1 + k0 + 64, 0, 1);
        __builtin_prefetch(w2 + k0 + 64, 0, 1);
        __builtin_prefetch(w3 + k0 + 64, 0, 1);
      }
    }
    acc0 = __builtin_amdgcn_wmma_f32_16x16x32_bf16(false, f[s].a.v, false, f[s].b0.v,
                                                   (short)0, acc0, false, false);
    acc1 = __builtin_amdgcn_wmma_f32_16x16x32_bf16(false, f[s].a.v, false, f[s].b1.v,
                                                   (short)0, acc1, false, false);
    acc2 = __builtin_amdgcn_wmma_f32_16x16x32_bf16(false, f[s].a.v, false, f[s].b2.v,
                                                   (short)0, acc2, false, false);
    acc3 = __builtin_amdgcn_wmma_f32_16x16x32_bf16(false, f[s].a.v, false, f[s].b3.v,
                                                   (short)0, acc3, false, false);
  }

  // C layout per tile: VGPR v, lanes 0-15 -> M = v; lanes 16-31 -> M = v + 8; N = n + rl
#pragma unroll
  for (int tn = 0; tn < 4; ++tn) {
    const v8f acc = (tn == 0) ? acc0 : (tn == 1) ? acc1 : (tn == 2) ? acc2 : acc3;
    const int col = n0 + tn * 16 + rl;
    const float bval = bias ? bias[col] : 0.0f;
#pragma unroll
    for (int v = 0; v < 8; ++v) {
      const int r = m0 + v + hi * 8;
      float val = acc[v] + bval;
      if (addend) val += addend[(size_t)r * N + col];
      C[(size_t)r * N + col] = val;
    }
  }
}

// ============================================================
// Elementwise / small kernels
// ============================================================
__global__ void cvt_f32_bf16_kernel(const float* __restrict__ in,
                                    __bf16* __restrict__ out, size_t n) {
  size_t i = (size_t)blockIdx.x * blockDim.x + threadIdx.x;
  if (i < n) out[i] = f32_to_bf16(in[i]);
}

__global__ void lengths_kernel(const unsigned char* __restrict__ mask,
                               int* __restrict__ lengths) {
  int b = threadIdx.x;
  if (b < BATCH) {
    int cnt = 0;
    for (int l = 0; l < SEQLEN; ++l) cnt += (mask[b * SEQLEN + l] == 0) ? 1 : 0;
    lengths[b] = cnt;
  }
}

__global__ void embed_kernel(const int* __restrict__ ids,
                             const float* __restrict__ emb,
                             float* __restrict__ X) {
  size_t i = (size_t)blockIdx.x * blockDim.x + threadIdx.x;
  if (i >= (size_t)BL * D_MODEL) return;
  int d = (int)(i % D_MODEL);
  int row = (int)(i / D_MODEL);
  int id = ids[row];
  X[i] = emb[(size_t)id * D_MODEL + d];
}

__global__ void flip_kernel(const float* __restrict__ in,
                            float* __restrict__ out,
                            const int* __restrict__ lengths) {
  size_t i = (size_t)blockIdx.x * blockDim.x + threadIdx.x;
  if (i >= (size_t)BL * D_MODEL) return;
  int d = (int)(i % D_MODEL);
  int l = (int)((i / D_MODEL) % SEQLEN);
  int b = (int)(i / ((size_t)D_MODEL * SEQLEN));
  int len = lengths[b];
  int src = (l < len) ? (len - 1 - l) : l;
  out[i] = in[((size_t)b * SEQLEN + src) * D_MODEL + d];
}

// depthwise causal conv (K=4) over seq + bias + silu, reading xBC slice of zxbcdt
__global__ void conv_silu_kernel(const float* __restrict__ zx,   // (BL, D_IN_PROJ)
                                 const float* __restrict__ cw,   // (CONV_DIM, D_CONV)
                                 const float* __restrict__ cb,   // (CONV_DIM)
                                 float* __restrict__ out) {      // (BL, CONV_DIM)
  size_t i = (size_t)blockIdx.x * blockDim.x + threadIdx.x;
  if (i >= (size_t)BL * CONV_DIM) return;
  int c = (int)(i % CONV_DIM);
  int l = (int)((i / CONV_DIM) % SEQLEN);
  int b = (int)(i / ((size_t)CONV_DIM * SEQLEN));
  float acc = cb[c];
#pragma unroll
  for (int k = 0; k < D_CONV; ++k) {
    int ls = l - (D_CONV - 1) + k;
    if (ls >= 0)
      acc += zx[((size_t)b * SEQLEN + ls) * D_IN_PROJ + D_INNER + c] * cw[c * D_CONV + k];
  }
  out[i] = siluf(acc);
}

// ============================================================
// SSD sequential scan. grid = BATCH*NHEADS blocks, 64 threads.
// Lane p keeps h[p, 0..63] in registers; B/C staged in LDS per step.
// ============================================================
__global__ __launch_bounds__(64)
void ssd_scan_kernel(const float* __restrict__ xbc,   // (BL, CONV_DIM) post conv+silu
                     const float* __restrict__ zx,    // (BL, D_IN_PROJ) for raw dt cols
                     const float* __restrict__ dtb,   // (NHEADS) layer slice
                     const float* __restrict__ alog,  // (NHEADS)
                     const float* __restrict__ dss,   // (NHEADS)
                     float* __restrict__ y) {         // (BL, D_INNER)
  const int b = blockIdx.x / NHEADS;
  const int h = blockIdx.x % NHEADS;
  const int p = threadIdx.x;                 // 0..63
  const float Aval = -__expf(alog[h]);
  const float Dval = dss[h];
  const float dtbias = dtb[h];

  float hs[D_STATE];
#pragma unroll
  for (int n = 0; n < D_STATE; ++n) hs[n] = 0.0f;

  __shared__ float Bs[D_STATE];
  __shared__ float Cs[D_STATE];

  for (int t = 0; t < SEQLEN; ++t) {
    const size_t rowi = (size_t)b * SEQLEN + t;
    const float* row = xbc + rowi * CONV_DIM;
    Bs[p] = row[D_INNER + p];
    Cs[p] = row[D_INNER + D_STATE + p];
    __syncthreads();

    const float dtraw = zx[rowi * D_IN_PROJ + D_INNER + CONV_DIM + h] + dtbias;
    const float dt = softplusf(dtraw);
    const float dA = __expf(dt * Aval);
    const float xv = row[h * HEADDIM + p];
    const float dtx = dt * xv;

    float acc = 0.0f;
#pragma unroll
    for (int n = 0; n < D_STATE; ++n) {
      hs[n] = hs[n] * dA + dtx * Bs[n];
      acc += hs[n] * Cs[n];
    }
    y[rowi * D_INNER + h * HEADDIM + p] = acc + xv * Dval;
    __syncthreads();
  }
}

// y_norm = rmsnorm(y * silu(z), norm_w)  -> bf16 for out_proj GEMM
__global__ __launch_bounds__(256)
void gate_rmsnorm_kernel(const float* __restrict__ y,   // (BL, D_INNER)
                         const float* __restrict__ zx,  // (BL, D_IN_PROJ), z = first D_INNER
                         const float* __restrict__ nw,  // (D_INNER)
                         __bf16* __restrict__ out) {    // (BL, D_INNER)
  const int row = blockIdx.x;
  const float* yr = y + (size_t)row * D_INNER;
  const float* zr = zx + (size_t)row * D_IN_PROJ;
  float vals[D_INNER / 256];
  float ss = 0.0f;
#pragma unroll
  for (int i = 0; i < D_INNER / 256; ++i) {
    int idx = threadIdx.x + i * 256;
    float v = yr[idx] * siluf(zr[idx]);
    vals[i] = v;
    ss += v * v;
  }
  __shared__ float red[256];
  red[threadIdx.x] = ss;
  __syncthreads();
  for (int s = 128; s > 0; s >>= 1) {
    if (threadIdx.x < s) red[threadIdx.x] += red[threadIdx.x + s];
    __syncthreads();
  }
  const float scale = rsqrtf(red[0] / (float)D_INNER + EPSF);
#pragma unroll
  for (int i = 0; i < D_INNER / 256; ++i) {
    int idx = threadIdx.x + i * 256;
    out[(size_t)row * D_INNER + idx] = f32_to_bf16(vals[i] * scale * nw[idx]);
  }
}

__global__ void concat_bf_kernel(const float* __restrict__ xf,
                                 const float* __restrict__ xb,
                                 __bf16* __restrict__ out) {   // (BL, 2*D_MODEL)
  size_t i = (size_t)blockIdx.x * blockDim.x + threadIdx.x;
  if (i >= (size_t)BL * 2 * D_MODEL) return;
  int d = (int)(i % (2 * D_MODEL));
  int row = (int)(i / (2 * D_MODEL));
  float v = (d < D_MODEL) ? xf[(size_t)row * D_MODEL + d]
                          : xb[(size_t)row * D_MODEL + (d - D_MODEL)];
  out[i] = f32_to_bf16(v);
}

__global__ void combine_kernel(const float* __restrict__ zg,  // (BL, D_MODEL) pre-sigmoid
                               const float* __restrict__ xf,
                               const float* __restrict__ xb,
                               float* __restrict__ X) {
  size_t i = (size_t)blockIdx.x * blockDim.x + threadIdx.x;
  if (i >= (size_t)BL * D_MODEL) return;
  float g = sigmf(zg[i]);
  X[i] = g * xf[i] + (1.0f - g) * xb[i];
}

__global__ __launch_bounds__(256)
void final_rmsnorm_kernel(const float* __restrict__ X,
                          const float* __restrict__ nw,
                          __bf16* __restrict__ out) {   // (BL, D_MODEL)
  const int row = blockIdx.x;
  const float* xr = X + (size_t)row * D_MODEL;
  float vals[D_MODEL / 256];
  float ss = 0.0f;
#pragma unroll
  for (int i = 0; i < D_MODEL / 256; ++i) {
    int idx = threadIdx.x + i * 256;
    float v = xr[idx];
    vals[i] = v;
    ss += v * v;
  }
  __shared__ float red[256];
  red[threadIdx.x] = ss;
  __syncthreads();
  for (int s = 128; s > 0; s >>= 1) {
    if (threadIdx.x < s) red[threadIdx.x] += red[threadIdx.x + s];
    __syncthreads();
  }
  const float scale = rsqrtf(red[0] / (float)D_MODEL + EPSF);
#pragma unroll
  for (int i = 0; i < D_MODEL / 256; ++i) {
    int idx = threadIdx.x + i * 256;
    out[(size_t)row * D_MODEL + idx] = f32_to_bf16(vals[i] * scale * nw[idx]);
  }
}

// ============================================================
// Host launcher
// ============================================================
static inline int nblk(size_t n, int b) { return (int)((n + b - 1) / b); }

extern "C" void kernel_launch(void* const* d_in, const int* in_sizes, int n_in,
                              void* d_out, int out_size, void* d_ws, size_t ws_size,
                              hipStream_t stream) {
  (void)in_sizes; (void)n_in; (void)out_size; (void)ws_size;
  const int*   ids        = (const int*)d_in[0];
  const unsigned char* mask = (const unsigned char*)d_in[1];
  const float* emb        = (const float*)d_in[2];
  const float* in_proj_w  = (const float*)d_in[3];   // (2, 2304, 512)
  const float* conv_w     = (const float*)d_in[4];   // (2, 1152, 4)
  const float* conv_b     = (const float*)d_in[5];   // (2, 1152)
  const float* dt_bias    = (const float*)d_in[6];   // (2, 16)
  const float* A_log      = (const float*)d_in[7];   // (2, 16)
  const float* D_ssm      = (const float*)d_in[8];   // (2, 16)
  const float* norm_w     = (const float*)d_in[9];   // (2, 1024)
  const float* out_proj_w = (const float*)d_in[10];  // (2, 512, 1024)
  const float* aggr_w     = (const float*)d_in[11];  // (512, 1024)
  const float* aggr_b     = (const float*)d_in[12];  // (512)
  const float* norm_f_w   = (const float*)d_in[13];  // (512)
  const float* lm_head_w  = (const float*)d_in[14];  // (32768, 512)
  const float* lm_head_b  = (const float*)d_in[15];  // (32768)
  float* out = (float*)d_out;                        // (2, 512, 32768)

  // ---- workspace bump allocator (total ~75 MiB) ----
  size_t off = 0;
  auto alloc = [&](size_t bytes) -> void* {
    void* p = (char*)d_ws + off;
    off += (bytes + 255) & ~(size_t)255;
    return p;
  };
  __bf16* wInBf   = (__bf16*)alloc((size_t)N_LAYERS * D_IN_PROJ * D_MODEL * 2);
  __bf16* wOutBf  = (__bf16*)alloc((size_t)N_LAYERS * D_MODEL * D_INNER * 2);
  __bf16* wAggrBf = (__bf16*)alloc((size_t)D_MODEL * 2 * D_MODEL * 2);
  __bf16* wLmBf   = (__bf16*)alloc((size_t)VOCABSZ * D_MODEL * 2);
  float*  X       = (float*)alloc((size_t)BL * D_MODEL * 4);
  float*  Xflip   = (float*)alloc((size_t)BL * D_MODEL * 4);
  __bf16* XdirBf  = (__bf16*)alloc((size_t)BL * D_MODEL * 2);
  float*  zx      = (float*)alloc((size_t)BL * D_IN_PROJ * 4);
  float*  xbcConv = (float*)alloc((size_t)BL * CONV_DIM * 4);
  float*  ybuf    = (float*)alloc((size_t)BL * D_INNER * 4);
  __bf16* ybf     = (__bf16*)alloc((size_t)BL * D_INNER * 2);
  float*  XfOut   = (float*)alloc((size_t)BL * D_MODEL * 4);
  float*  XbTmp   = (float*)alloc((size_t)BL * D_MODEL * 4);
  float*  XbOut   = (float*)alloc((size_t)BL * D_MODEL * 4);
  __bf16* catBf   = (__bf16*)alloc((size_t)BL * 2 * D_MODEL * 2);
  float*  zg      = (float*)alloc((size_t)BL * D_MODEL * 4);
  __bf16* hBf     = (__bf16*)alloc((size_t)BL * D_MODEL * 2);
  int*    lens    = (int*)alloc(BATCH * sizeof(int));

  auto cvt = [&](const float* src, __bf16* dst, size_t n) {
    cvt_f32_bf16_kernel<<<nblk(n, 256), 256, 0, stream>>>(src, dst, n);
  };
  // K is compile-time: only 512 and 1024 occur in this model
  auto gemm = [&](const __bf16* A, const __bf16* W, float* C,
                  const float* bias, const float* addend, int M, int N, int K) {
    dim3 grid(M / GEMM_BM, N / GEMM_BN);
    if (K == 512)
      gemm_bf16_wmma<512><<<grid, 256, 0, stream>>>(A, W, C, bias, addend, M, N);
    else
      gemm_bf16_wmma<1024><<<grid, 256, 0, stream>>>(A, W, C, bias, addend, M, N);
  };

  // weight conversion (f32 -> bf16, once per launch; deterministic)
  cvt(in_proj_w,  wInBf,   (size_t)N_LAYERS * D_IN_PROJ * D_MODEL);
  cvt(out_proj_w, wOutBf,  (size_t)N_LAYERS * D_MODEL * D_INNER);
  cvt(aggr_w,     wAggrBf, (size_t)D_MODEL * 2 * D_MODEL);
  cvt(lm_head_w,  wLmBf,   (size_t)VOCABSZ * D_MODEL);

  lengths_kernel<<<1, 64, 0, stream>>>(mask, lens);
  embed_kernel<<<nblk((size_t)BL * D_MODEL, 256), 256, 0, stream>>>(ids, emb, X);

  // one Mamba2 direction: Xin (f32 BL x D_MODEL) -> Xout = block(Xin)
  auto run_dir = [&](const float* Xin, float* Xout, int l) {
    cvt(Xin, XdirBf, (size_t)BL * D_MODEL);
    gemm(XdirBf, wInBf + (size_t)l * D_IN_PROJ * D_MODEL, zx,
         nullptr, nullptr, BL, D_IN_PROJ, D_MODEL);
    conv_silu_kernel<<<nblk((size_t)BL * CONV_DIM, 256), 256, 0, stream>>>(
        zx, conv_w + (size_t)l * CONV_DIM * D_CONV, conv_b + (size_t)l * CONV_DIM, xbcConv);
    ssd_scan_kernel<<<BATCH * NHEADS, 64, 0, stream>>>(
        xbcConv, zx, dt_bias + l * NHEADS, A_log + l * NHEADS, D_ssm + l * NHEADS, ybuf);
    gate_rmsnorm_kernel<<<BL, 256, 0, stream>>>(ybuf, zx, norm_w + (size_t)l * D_INNER, ybf);
    gemm(ybf, wOutBf + (size_t)l * D_MODEL * D_INNER, Xout,
         nullptr, /*residual*/ Xin, BL, D_MODEL, D_INNER);
  };

  for (int l = 0; l < N_LAYERS; ++l) {
    flip_kernel<<<nblk((size_t)BL * D_MODEL, 256), 256, 0, stream>>>(X, Xflip, lens);
    run_dir(X, XfOut, l);
    run_dir(Xflip, XbTmp, l);
    flip_kernel<<<nblk((size_t)BL * D_MODEL, 256), 256, 0, stream>>>(XbTmp, XbOut, lens);

    concat_bf_kernel<<<nblk((size_t)BL * 2 * D_MODEL, 256), 256, 0, stream>>>(XfOut, XbOut, catBf);
    gemm(catBf, wAggrBf, zg, aggr_b, nullptr, BL, D_MODEL, 2 * D_MODEL);
    combine_kernel<<<nblk((size_t)BL * D_MODEL, 256), 256, 0, stream>>>(zg, XfOut, XbOut, X);
  }

  final_rmsnorm_kernel<<<BL, 256, 0, stream>>>(X, norm_f_w, hBf);
  // logits: (1024, 32768) = h (1024,512) x lm_head_w (32768,512)^T + b
  gemm(hBf, wLmBf, out, lm_head_b, nullptr, BL, VOCABSZ, D_MODEL);
}
